// VideoRoPESelfAttention_86328842649968
// MI455X (gfx1250) — matrix operand order repaired and checked by reading, not compiled, not run on hardware
//
#include <hip/hip_runtime.h>
#include <hip/hip_bf16.h>
#include <math.h>

typedef __attribute__((ext_vector_type(16))) __bf16 v16bf;
typedef __attribute__((ext_vector_type(8)))  __bf16 bf16x8;
typedef __attribute__((ext_vector_type(8)))  float  v8f;
typedef __attribute__((ext_vector_type(4)))  int    v4i;

#define WMMA_BF16(a, b, c) \
  __builtin_amdgcn_wmma_f32_16x16x32_bf16(false, (a), false, (b), (short)0, (c), false, false)

// Async global->LDS copy (CDNA5 GLOBAL_LOAD_ASYNC_TO_LDS_B128, ASYNCcnt).
// Probe-confirmed: 4 args, generic int4* pointers (global src, lds dst, imm
// offset, imm cpol). Guarded: host pass / older toolchains fall back to a
// register bounce.
#if defined(__has_builtin)
#if __has_builtin(__builtin_amdgcn_global_load_async_to_lds_b128) && \
    __has_builtin(__builtin_amdgcn_s_wait_asynccnt)
#define HAVE_ASYNC 1
#endif
#endif
#ifndef HAVE_ASYNC
#define HAVE_ASYNC 0
#endif

#if HAVE_ASYNC
#define ASYNC_CP16(l, g) \
  __builtin_amdgcn_global_load_async_to_lds_b128((v4i*)(g), (v4i*)(l), 0, 0)
#define ASYNC_WAIT() __builtin_amdgcn_s_wait_asynccnt(0)
#else
#define ASYNC_CP16(l, g) (*(bf16x8*)(l) = *(const bf16x8*)(g))
#define ASYNC_WAIT()
#endif

constexpr int Bb   = 4;
constexpr int Nn   = 1576;          // tokens per batch (8 frames * 197)
constexpr int Dd   = 768;
constexpr int Hh   = 12;
constexpr int HD   = 64;            // head dim
constexpr int TPF  = 197;
constexpr int DPC  = 256;           // rope channels per third
constexpr int Mrows = Bb * Nn;      // 6304 = 16*394
constexpr int Npad  = 1600;         // padded seq len (multiple of 32)
constexpr int QKVC  = 3 * Dd;       // 2304
constexpr int QTILES = 99;          // ceil(1576/16)
constexpr int QGRP   = 13;          // ceil(99/8) query-tile groups per (b,h)

// Concatenate two 8-element chunks into a 16-element WMMA fragment.
__device__ inline v16bf cat8(bf16x8 lo, bf16x8 hi) {
  v16bf r;
#pragma unroll
  for (int e = 0; e < 8; ++e) { r[e] = lo[e]; r[8 + e] = hi[e]; }
  return r;
}

// A fragment (16x32, bf16) from a row-major row pointer at K-chunk base.
// Lane layout: lane<16 -> K {0..15}, lane>=16 -> K {16..31}.
__device__ inline v16bf loadA(const __bf16* rowk0, int hiA) {
  bf16x8 lo = *(const bf16x8*)(rowk0 + 2 * hiA);
  bf16x8 hi = *(const bf16x8*)(rowk0 + 2 * hiA + 8);
  return cat8(lo, hi);
}

// ---------------------------------------------------------------------------
// RoPE + f32 -> bf16 conversion. One block per row (B*N rows), 256 threads.
// ---------------------------------------------------------------------------
__global__ __launch_bounds__(256) void rope_to_bf16(const float* __restrict__ x,
                                                    __bf16* __restrict__ xb) {
  int row = blockIdx.x;          // 0..6303
  int j   = threadIdx.x;         // 0..255
  int n   = row % Nn;
  int p   = n % TPF;
  float shifted = (p > 0) ? (float)(p - 1) : 0.f;
  int   fidx = j & 127;                                     // emb[j] = ang[j%128]
  float freq = __expf(-logf(10000.f) * (float)(2 * fidx) / (float)DPC);
  float ang  = shifted * freq;
  float c = __cosf(ang);
  float s = (p > 0) ? __sinf(ang) : 0.f;
  const float* xr = x  + (size_t)row * Dd;
  __bf16*      xo = xb + (size_t)row * Dd;
  float x1 = xr[j], x2 = xr[DPC + j], rest = xr[2 * DPC + j];
  xo[j]           = (__bf16)(x1 * c - x2 * s);
  xo[DPC + j]     = (__bf16)(x2 * c + x1 * s);
  xo[2 * DPC + j] = (__bf16)rest;
}

// ---------------------------------------------------------------------------
// Repack f32 weight [K][C] into WMMA B-fragment tiles:
//   Wf[((kt*NT + nt)*32 + lane)*16 + i] = W[kt*32 + (lane<16?0:16) + i][nt*16 + lane%16]
// so each lane's fragment is 32 contiguous bytes (two global_load_b128).
// ---------------------------------------------------------------------------
__global__ __launch_bounds__(256) void pack_weight_frag(const float* __restrict__ W,
                                                        __bf16* __restrict__ Wf,
                                                        int K, int C) {
  int o = blockIdx.x * blockDim.x + threadIdx.x;
  if (o >= K * C) return;
  int NT   = C / 16;
  int i    = o & 15;
  int lane = (o >> 4) & 31;
  int t    = o >> 9;              // (kt*NT + nt)
  int nt   = t % NT;
  int kt   = t / NT;
  int k = kt * 32 + ((lane < 16) ? 0 : 16) + i;
  int c = nt * 16 + (lane & 15);
  Wf[(size_t)o] = (__bf16)W[(size_t)k * C + c];
}

__global__ __launch_bounds__(256) void zero_ws(unsigned int* __restrict__ p, size_t ndw) {
  size_t i = (size_t)blockIdx.x * blockDim.x + threadIdx.x;
  size_t stride = (size_t)gridDim.x * blockDim.x;
  for (; i < ndw; i += stride) p[i] = 0u;
}

// ---------------------------------------------------------------------------
// QKV GEMM: [6304 x 768]bf16 @ [768 x 2304]bf16 + bias. One wave -> 16x128
// strip (8 accumulators). A and B fragments are software-pipelined so the
// wmma never waits on the load it just issued. Epilogue scatters q,k
// row-major [b][h][tok][64] and v transposed [b][h][64][tok].
// ---------------------------------------------------------------------------
__global__ __launch_bounds__(256) void qkv_gemm(const __bf16* __restrict__ Xb,
                                                const __bf16* __restrict__ Wf,
                                                const float*  __restrict__ bias,
                                                __bf16* __restrict__ qw,
                                                __bf16* __restrict__ kw,
                                                __bf16* __restrict__ vtw) {
  const int strips   = QKVC / 128;       // 18
  const int rowTiles = Mrows / 16;       // 394
  const int NT       = QKVC / 16;        // 144
  const int KT       = Dd / 32;          // 24
  int lane = threadIdx.x & 31;
  int wid  = blockIdx.x * (blockDim.x >> 5) + (threadIdx.x >> 5);
  if (wid >= rowTiles * strips) return;
  int rt = wid / strips, cs = wid % strips;

  int lhalf = lane & 15;
  int hiA = (lane < 16) ? 0 : 8;         // A-layout K base / C-layout M base
  int mrow = rt * 16 + lhalf;
  const __bf16* arow = Xb + (size_t)mrow * Dd;

  v8f acc[8] = {};
  v16bf a = loadA(arow, hiA);
  for (int kt = 0; kt < KT; ++kt) {
    const __bf16* wptr = Wf + ((size_t)(kt * NT + cs * 8) * 32 + lane) * 16;
    __builtin_prefetch(wptr + (size_t)NT * 512, 0, 1);   // next K-chunk tile
    v16bf anext = a;
    if (kt + 1 < KT) anext = loadA(arow + (kt + 1) * 32, hiA);
    v16bf bcur = *(const v16bf*)(wptr);
#pragma unroll
    for (int bg = 0; bg < 8; ++bg) {
      v16bf bnext = bcur;
      if (bg < 7) bnext = *(const v16bf*)(wptr + (bg + 1) * 512);
      acc[bg] = WMMA_BF16(a, bcur, acc[bg]);
      bcur = bnext;
    }
    a = anext;
  }

  // Hoist the div/mod-heavy address math out of the 64-store loop.
  size_t baseQK[8], baseV[8];
#pragma unroll
  for (int r = 0; r < 8; ++r) {
    int m  = rt * 16 + r + hiA;          // C layout: M = r (+8 for upper half)
    int b_ = m / Nn, n = m % Nn;
    baseQK[r] = ((size_t)b_ * Hh * Npad + n) * HD;
    baseV[r]  = (size_t)b_ * Hh * HD * Npad + n;
  }
#pragma unroll
  for (int bg = 0; bg < 8; ++bg) {
    int c  = cs * 128 + bg * 16 + lhalf;
    float bv = bias[c];
    int sec = c / Dd;                    // 0=q 1=k 2=v (uniform per strip)
    int cc  = c % Dd;
    int h = cc / HD, dm = cc % HD;
    size_t qkoff = (size_t)h * Npad * HD + dm;
    size_t voff  = ((size_t)h * HD + dm) * Npad;
#pragma unroll
    for (int r = 0; r < 8; ++r) {
      __bf16 bfv = (__bf16)(acc[bg][r] + bv);
      if (sec == 0)      qw [baseQK[r] + qkoff] = bfv;
      else if (sec == 1) kw [baseQK[r] + qkoff] = bfv;
      else               vtw[baseV[r] + voff]   = bfv;
    }
  }
}

// ---------------------------------------------------------------------------
// Flash-style attention, cooperative K/V staging.
// One block = 8 waves sharing one (b,h); wave wv owns query tile qg*8+wv.
// Each 32-key step, the block async-copies the K tile (32x64 bf16, 4KB) and
// V tile (64x32 bf16, 4KB) into double-buffered LDS (ASYNCcnt), prefetching
// step t+1 while computing step t. Probs go through a per-wave LDS tile to
// convert C-layout(f32) -> A-layout(bf16).
// ---------------------------------------------------------------------------
__global__ __launch_bounds__(256) void attention(const __bf16* __restrict__ qw,
                                                 const __bf16* __restrict__ kw,
                                                 const __bf16* __restrict__ vtw,
                                                 __bf16* __restrict__ ctx) {
  __shared__ __bf16 kbuf[2][32][64];     // [buf][token][dim]   4KB each
  __shared__ __bf16 vbuf[2][64][32];     // [buf][dim][key]     4KB each
  __shared__ __bf16 pbuf[8][16 * 32];    // per-wave probs tile 8KB

  int tid   = threadIdx.x;
  int lane  = tid & 31;
  int wv    = tid >> 5;
  int bh    = blockIdx.x / QGRP;         // b*H + h
  int qg    = blockIdx.x % QGRP;
  int qtRaw = qg * 8 + wv;
  bool valid = qtRaw < QTILES;
  int qt = valid ? qtRaw : (QTILES - 1);

  const __bf16* qb = qw  + (size_t)bh * Npad * HD;
  const __bf16* kb = kw  + (size_t)bh * Npad * HD;
  const __bf16* vb = vtw + (size_t)bh * HD * Npad;

  // Cooperative copy slots: 16B per thread per buffer.
  int kt_ = tid >> 3, ksg = tid & 7;         // k: token, 8B-seg of 64 dims
  int vd_ = tid >> 2, vsg = tid & 3;         // v: dim,   8B-seg of 32 keys
  const __bf16* ksrc = kb + (size_t)kt_ * HD + ksg * 8;   // + key0*HD
  const __bf16* vsrc = vb + (size_t)vd_ * Npad + vsg * 8; // + key0
  __bf16* kdst0 = &kbuf[0][kt_][ksg * 8];
  __bf16* kdst1 = &kbuf[1][kt_][ksg * 8];
  __bf16* vdst0 = &vbuf[0][vd_][vsg * 8];
  __bf16* vdst1 = &vbuf[1][vd_][vsg * 8];

  int lhalf = lane & 15;
  int hiA = (lane < 16) ? 0 : 8;
  int hiB = (lane < 16) ? 0 : 16;

  // Q fragments: 16 x 64 = two 16x32 A tiles (4x b128 total).
  int qrow = qt * 16 + lhalf;            // <= 1583 < Npad
  const __bf16* qrp = qb + (size_t)qrow * HD;
  v16bf aq0 = loadA(qrp, hiA);
  v16bf aq1 = loadA(qrp + 32, hiA);

  float mrow[8], lrow[8];
  v8f acc[4] = {};
#pragma unroll
  for (int r = 0; r < 8; ++r) { mrow[r] = -3.0e38f; lrow[r] = 0.f; }

  // Prologue: stage key tile 0.
  ASYNC_CP16(kdst0, ksrc);
  ASYNC_CP16(vdst0, vsrc);
  ASYNC_WAIT();
  __syncthreads();

  for (int kt = 0; kt < Npad / 32; ++kt) {      // 50 iterations of 32 keys
    int key0 = kt * 32;
    int cur  = kt & 1;
    if (kt + 1 < Npad / 32) {                   // stage next tile (async)
      int nk = (kt + 1) * 32;
      ASYNC_CP16(cur ? kdst0 : kdst1, ksrc + (size_t)nk * HD);
      ASYNC_CP16(cur ? vdst0 : vdst1, vsrc + nk);
    }

    const __bf16* kt_lds = &kbuf[cur][0][0];
    const __bf16* vt_lds = &vbuf[cur][0][0];
    v8f s0 = {}, s1 = {};
#pragma unroll
    for (int ch = 0; ch < 2; ++ch) {            // head-dim chunks K=0..31,32..63
      v16bf b0 = *(const v16bf*)(kt_lds + (size_t)lhalf * HD + ch * 32 + hiB);
      v16bf b1 = *(const v16bf*)(kt_lds + (size_t)(16 + lhalf) * HD + ch * 32 + hiB);
      const v16bf aq = ch ? aq1 : aq0;
      s0 = WMMA_BF16(aq, b0, s0);
      s1 = WMMA_BF16(aq, b1, s1);
    }

    // Online softmax update (row = r + hiA lives across one 16-lane half).
    int keyA = key0 + lhalf, keyB = key0 + 16 + lhalf;
    float sc[8];
#pragma unroll
    for (int r = 0; r < 8; ++r) {
      float v0 = (keyA < Nn) ? s0[r] * 0.125f : -3.0e38f;   // 1/sqrt(64)
      float v1 = (keyB < Nn) ? s1[r] * 0.125f : -3.0e38f;
      float t = fmaxf(v0, v1);
      t = fmaxf(t, __shfl_xor(t, 1, 32));
      t = fmaxf(t, __shfl_xor(t, 2, 32));
      t = fmaxf(t, __shfl_xor(t, 4, 32));
      t = fmaxf(t, __shfl_xor(t, 8, 32));
      float mn = fmaxf(mrow[r], t);
      float p0 = __expf(v0 - mn);
      float p1 = __expf(v1 - mn);
      float scl = __expf(mrow[r] - mn);
      float rs = p0 + p1;
      rs += __shfl_xor(rs, 1, 32);
      rs += __shfl_xor(rs, 2, 32);
      rs += __shfl_xor(rs, 4, 32);
      rs += __shfl_xor(rs, 8, 32);
      lrow[r] = lrow[r] * scl + rs;
      mrow[r] = mn;
      sc[r]   = scl;
      pbuf[wv][(r + hiA) * 32 + lhalf]      = (__bf16)p0;
      pbuf[wv][(r + hiA) * 32 + 16 + lhalf] = (__bf16)p1;
    }
#pragma unroll
    for (int bg = 0; bg < 4; ++bg)
#pragma unroll
      for (int r = 0; r < 8; ++r) acc[bg][r] *= sc[r];

    __builtin_amdgcn_wave_barrier();   // per-wave LDS region; DS ops in-order

    // Probs 16x32 A fragment from LDS (2x ds_load_b128).
    bf16x8 plo = *(const bf16x8*)(&pbuf[wv][lhalf * 32 + 2 * hiA]);
    bf16x8 phi = *(const bf16x8*)(&pbuf[wv][lhalf * 32 + 2 * hiA + 8]);
    v16bf ap = cat8(plo, phi);

    // acc += P @ V from the staged V tile ([dim][key] -> contiguous frags).
#pragma unroll
    for (int bg = 0; bg < 4; ++bg) {
      int dim = bg * 16 + lhalf;
      v16bf bvf = *(const v16bf*)(vt_lds + (size_t)dim * 32 + hiB);
      acc[bg] = WMMA_BF16(ap, bvf, acc[bg]);
    }

    ASYNC_WAIT();        // next tile landed (this wave's async ops)
    __syncthreads();     // all waves done reading cur + writing next
  }

  // Normalize and store ctx [b][n][h*64+dim] in bf16.
  int b_ = bh / Hh, h = bh % Hh;
#pragma unroll
  for (int r = 0; r < 8; ++r) {
    float inv = 1.f / lrow[r];
    int n = qt * 16 + r + hiA;
    if (valid && n < Nn) {
#pragma unroll
      for (int bg = 0; bg < 4; ++bg) {
        int col = h * 64 + bg * 16 + lhalf;
        ctx[((size_t)(b_ * Nn + n)) * Dd + col] = (__bf16)(acc[bg][r] * inv);
      }
    }
  }
}

// ---------------------------------------------------------------------------
// Output projection: ctx[6304x768]bf16 @ Wout[768x768]bf16 + bout -> f32 out.
// Same pipelined 16x128-strip structure as qkv_gemm.
// ---------------------------------------------------------------------------
__global__ __launch_bounds__(256) void out_gemm(const __bf16* __restrict__ Cx,
                                                const __bf16* __restrict__ Wf,
                                                const float*  __restrict__ bias,
                                                float* __restrict__ out) {
  const int strips   = Dd / 128;         // 6
  const int rowTiles = Mrows / 16;       // 394
  const int NT       = Dd / 16;          // 48
  const int KT       = Dd / 32;          // 24
  int lane = threadIdx.x & 31;
  int wid  = blockIdx.x * (blockDim.x >> 5) + (threadIdx.x >> 5);
  if (wid >= rowTiles * strips) return;
  int rt = wid / strips, cs = wid % strips;

  int lhalf = lane & 15;
  int hiA = (lane < 16) ? 0 : 8;
  int mrow = rt * 16 + lhalf;
  const __bf16* arow = Cx + (size_t)mrow * Dd;

  v8f acc[8] = {};
  v16bf a = loadA(arow, hiA);
  for (int kt = 0; kt < KT; ++kt) {
    const __bf16* wptr = Wf + ((size_t)(kt * NT + cs * 8) * 32 + lane) * 16;
    __builtin_prefetch(wptr + (size_t)NT * 512, 0, 1);
    v16bf anext = a;
    if (kt + 1 < KT) anext = loadA(arow + (kt + 1) * 32, hiA);
    v16bf bcur = *(const v16bf*)(wptr);
#pragma unroll
    for (int bg = 0; bg < 8; ++bg) {
      v16bf bnext = bcur;
      if (bg < 7) bnext = *(const v16bf*)(wptr + (bg + 1) * 512);
      acc[bg] = WMMA_BF16(a, bcur, acc[bg]);
      bcur = bnext;
    }
    a = anext;
  }
#pragma unroll
  for (int bg = 0; bg < 8; ++bg) {
    int c = cs * 128 + bg * 16 + lhalf;
    float bv = bias[c];
#pragma unroll
    for (int r = 0; r < 8; ++r) {
      int m = rt * 16 + r + hiA;
      out[(size_t)m * Dd + c] = acc[bg][r] + bv;
    }
  }
}

// ---------------------------------------------------------------------------
extern "C" void kernel_launch(void* const* d_in, const int* in_sizes, int n_in,
                              void* d_out, int out_size, void* d_ws, size_t ws_size,
                              hipStream_t stream) {
  const float* x    = (const float*)d_in[0];   // [4,1576,768]
  const float* Wqkv = (const float*)d_in[1];   // [768,2304]
  const float* bqkv = (const float*)d_in[2];   // [2304]
  const float* Wout = (const float*)d_in[3];   // [768,768]
  const float* bout = (const float*)d_in[4];   // [768]
  float* out = (float*)d_out;

  char* wsp = (char*)d_ws;
  auto alloc = [&](size_t bytes) -> char* {
    char* p = wsp;
    wsp += (bytes + 255) & ~(size_t)255;
    return p;
  };
  __bf16* xb    = (__bf16*)alloc((size_t)Mrows * Dd * 2);
  __bf16* wqkvf = (__bf16*)alloc((size_t)Dd * QKVC * 2);
  __bf16* woutf = (__bf16*)alloc((size_t)Dd * Dd * 2);
  size_t qkvBytes = (size_t)Bb * Hh * Npad * HD * 2;   // 9,830,400 (mult of 256)
  __bf16* qw  = (__bf16*)alloc(qkvBytes);
  __bf16* kw  = (__bf16*)alloc(qkvBytes);
  __bf16* vtw = (__bf16*)alloc(qkvBytes);
  __bf16* ctx = (__bf16*)alloc((size_t)Mrows * Dd * 2);

  rope_to_bf16<<<Mrows, 256, 0, stream>>>(x, xb);
  pack_weight_frag<<<(Dd * QKVC + 255) / 256, 256, 0, stream>>>(Wqkv, wqkvf, Dd, QKVC);
  pack_weight_frag<<<(Dd * Dd + 255) / 256, 256, 0, stream>>>(Wout, woutf, Dd, Dd);
  zero_ws<<<2048, 256, 0, stream>>>((unsigned int*)qw, (3 * qkvBytes) / 4);

  int qkvWaves = (Mrows / 16) * (QKVC / 128);          // 7092
  qkv_gemm<<<(qkvWaves + 7) / 8, 256, 0, stream>>>(xb, wqkvf, bqkv, qw, kw, vtw);

  int attBlocks = Bb * Hh * QGRP;                      // 624
  attention<<<attBlocks, 256, 0, stream>>>(qw, kw, vtw, ctx);

  int outWaves = (Mrows / 16) * (Dd / 128);            // 2364
  out_gemm<<<(outWaves + 7) / 8, 256, 0, stream>>>(ctx, woutf, bout, out);
}
